// AdditiveAttention_80470507258106
// MI455X (gfx1250) — matrix operand order repaired and checked by reading, not compiled, not run on hardware
//
#include <hip/hip_runtime.h>
#include <hip/hip_bf16.h>

// Shapes from the reference
#define BB   8
#define LQ   256
#define LK   256
#define DQ   512
#define DV   512
#define UNI  256

typedef __attribute__((ext_vector_type(16))) __bf16 v16bf;
typedef __attribute__((ext_vector_type(8)))  __bf16 v8bf;
typedef __attribute__((ext_vector_type(4)))  __bf16 v4bf;
typedef __attribute__((ext_vector_type(8)))  float  v8f;

__device__ __forceinline__ v16bf cat8(v8bf lo, v8bf hi) {
    return __builtin_shufflevector(lo, hi, 0, 1, 2, 3, 4, 5, 6, 7,
                                   8, 9, 10, 11, 12, 13, 14, 15);
}

// ---------------------------------------------------------------------------
// fast tanh: native v_tanh_f32 on gfx1250 if available
// ---------------------------------------------------------------------------
__device__ __forceinline__ float fast_tanh(float x) {
#if __has_builtin(__builtin_amdgcn_tanhf)
    return __builtin_amdgcn_tanhf(x);
#else
    float t = __expf(2.0f * x);
    return (t - 1.0f) * __builtin_amdgcn_rcpf(t + 1.0f);
#endif
}

// ---------------------------------------------------------------------------
// gfx1250 async global->LDS copy (ASYNCcnt path).  LDS byte offset is the
// low 32 bits of the generic address of a __shared__ object.
// ---------------------------------------------------------------------------
__device__ __forceinline__ void async_g2l_b128(const void* g, void* lds) {
    unsigned l = (unsigned)(size_t)lds;
    asm volatile("global_load_async_to_lds_b128 %0, %1, off"
                 :: "v"(l), "v"(g) : "memory");
}
__device__ __forceinline__ void wait_async0() {
    asm volatile("s_wait_asynccnt 0x0" ::: "memory");
}

// ---------------------------------------------------------------------------
// bf16 WMMA fragment loads from pre-staged bf16 buffers (wave32 layouts).
// A (16x32, row-major):  lane m=l&15, h=l>>4; elements e=0..15 -> K=k0+16h+e
// B (32x16, stored transposed N-major): lane n=l&15, h=l>>4;
//                        elements e=0..15 -> K=k0+16h+e  (32 contiguous bytes)
// C/D f32: vgpr r holds row m=r+8h, col n=l&15
// ---------------------------------------------------------------------------
__device__ __forceinline__ v16bf load_a_frag(const __bf16* __restrict__ arow,
                                             int k0, int h) {
    v8bf lo = *(const v8bf*)(arow + k0 + 16 * h);
    v8bf hi = *(const v8bf*)(arow + k0 + 16 * h + 8);
    return cat8(lo, hi);
}
__device__ __forceinline__ v16bf load_b_frag(const __bf16* __restrict__ brow,
                                             int k0, int h) {
    return *(const v16bf*)(brow + k0 + 16 * h);   // 2x b128
}

// ---------------------------------------------------------------------------
// Staging pass 1: elementwise f32 -> bf16 (row-major A operands)
// ---------------------------------------------------------------------------
__global__ __launch_bounds__(256) void cvt_bf16_kernel(
    const float* __restrict__ in, __bf16* __restrict__ out, int n) {
    int i = (blockIdx.x * 256 + threadIdx.x) * 4;
    if (i < n) {
        float4 f = *(const float4*)(in + i);
        v4bf o = { (__bf16)f.x, (__bf16)f.y, (__bf16)f.z, (__bf16)f.w };
        *(v4bf*)(out + i) = o;
    }
}

// ---------------------------------------------------------------------------
// Staging pass 2: LDS-tiled transpose + convert.  in f32 [R x C] -> out bf16
// [C x R], one 32x32 tile per block, batched over blockIdx.z.
// ---------------------------------------------------------------------------
__global__ __launch_bounds__(256) void transpose_cvt_kernel(
    const float* __restrict__ in, __bf16* __restrict__ out, int R, int C) {
    __shared__ float tile[32][33];
    long base = (long)blockIdx.z * R * C;
    int c0 = blockIdx.x * 32, r0 = blockIdx.y * 32;
    int tx = threadIdx.x & 31, ty = threadIdx.x >> 5;   // 32 x 8
#pragma unroll
    for (int i = 0; i < 32; i += 8)
        tile[ty + i][tx] = in[base + (long)(r0 + ty + i) * C + c0 + tx];
    __syncthreads();
    __bf16* ob = out + (long)blockIdx.z * R * C;
#pragma unroll
    for (int i = 0; i < 32; i += 8)
        ob[(long)(c0 + ty + i) * R + r0 + tx] = (__bf16)tile[tx][ty + i];
}

// ---------------------------------------------------------------------------
// Kernel 1: qp = query @ Wq ; kp = key @ Wk   (bf16 WMMA, f32 accumulate)
//   One wave per 16x16 tile; K=512 fully unrolled -> 16 WMMAs/wave.
//   Per K-step: 4x global_load_b128 + 1x v_wmma_f32_16x16x32_bf16.
// ---------------------------------------------------------------------------
__global__ __launch_bounds__(256) void qk_proj_kernel(
    const __bf16* __restrict__ qbf, const __bf16* __restrict__ kbf,
    const __bf16* __restrict__ WqT, const __bf16* __restrict__ WkT,
    float* __restrict__ qp, float* __restrict__ kp) {
    int wave = (blockIdx.x * 256 + threadIdx.x) >> 5;
    int lane = threadIdx.x & 31;
    int h = lane >> 4, ln = lane & 15;

    const __bf16 *X, *Wt; float* P; int t;
    if (wave < 2048) { X = qbf; Wt = WqT; P = qp; t = wave; }
    else             { X = kbf; Wt = WkT; P = kp; t = wave - 2048; }
    int tm = t >> 4, tn = t & 15;

    const __bf16* arow = X + (long)(tm * 16 + ln) * DQ;   // A row, row-major
    const __bf16* brow = Wt + (long)(tn * 16 + ln) * DQ;  // B col, N-major

    v8f acc = {};
#pragma unroll
    for (int k0 = 0; k0 < DQ; k0 += 32) {
        v16bf a = load_a_frag(arow, k0, h);
        v16bf b = load_b_frag(brow, k0, h);
        acc = __builtin_amdgcn_wmma_f32_16x16x32_bf16(
            false, a, false, b, (short)0, acc, false, false);
    }
#pragma unroll
    for (int r = 0; r < 8; ++r)
        P[(long)(tm * 16 + r + 8 * h) * UNI + tn * 16 + ln] = acc[r];
}

// ---------------------------------------------------------------------------
// Kernel 2: additive score + mask + softmax (VALU/tanh bound).
//   k-tiles staged via async global->LDS (ASYNCcnt), next tile prefetched.
//   Also emits a bf16 copy of attn for the context GEMM's A operand.
// ---------------------------------------------------------------------------
__global__ __launch_bounds__(256) void score_softmax_kernel(
    const float* __restrict__ qp, const float* __restrict__ kp,
    const float* __restrict__ Wv, const unsigned char* __restrict__ mask,
    float* __restrict__ attn, __bf16* __restrict__ attn_bf) {
    __shared__ float qL[16 * 260];
    __shared__ float kL[16 * 260];
    __shared__ float wvL[UNI];
    __shared__ float sL[16 * 257];

    int b   = blockIdx.x >> 4;
    int q0  = (blockIdx.x & 15) * 16;
    int tid = threadIdx.x;

    wvL[tid] = Wv[tid];
    const float* qrow = qp + (long)(b * LQ + q0) * UNI;
#pragma unroll
    for (int i = tid; i < 16 * UNI; i += 256)
        qL[(i >> 8) * 260 + (i & 255)] = qrow[i];

    int qi = tid >> 4, ki = tid & 15;
    int sr = tid >> 4, sc = tid & 15;   // staging: row, chunk-lane

    for (int kt = 0; kt < 16; ++kt) {
        __syncthreads();   // previous kL readers done / qL visible (1st iter)
        const float* krow = kp + (long)(b * LK + kt * 16) * UNI;
        // async-copy this 16x256 f32 tile into padded LDS (4 x b128 / thread)
#pragma unroll
        for (int it = 0; it < 4; ++it) {
            int ch = sc + it * 16;                       // 64 chunks per row
            async_g2l_b128(krow + (long)sr * UNI + ch * 4,
                           &kL[sr * 260 + ch * 4]);
        }
        if (kt + 1 < 16) {   // prefetch next tile into cache
            const float* nxt = kp + (long)(b * LK + (kt + 1) * 16) * UNI;
            __builtin_prefetch(nxt + (tid << 4), 0, 1);
        }
        wait_async0();
        __syncthreads();

        const float* qr = &qL[qi * 260];
        const float* kr = &kL[ki * 260];
        float s = 0.0f;
#pragma unroll 8
        for (int u = 0; u < UNI; ++u)
            s = fmaf(wvL[u], fast_tanh(qr[u] + kr[u]), s);

        int kk = kt * 16 + ki;
        unsigned char m = mask[((long)(b * LQ + q0 + qi)) * LK + kk];
        sL[qi * 257 + kk] = m ? s : -1.0e6f;
    }
    __syncthreads();

    // softmax: 16 lanes per query row (rows align with half-wave groups)
    int r = tid >> 4, j = tid & 15;
    float vals[16];
    float mx = -3.4e38f;
#pragma unroll
    for (int i = 0; i < 16; ++i) {
        vals[i] = sL[r * 257 + j + 16 * i];
        mx = fmaxf(mx, vals[i]);
    }
#pragma unroll
    for (int off = 8; off; off >>= 1) mx = fmaxf(mx, __shfl_xor(mx, off, 16));
    float sum = 0.0f;
#pragma unroll
    for (int i = 0; i < 16; ++i) {
        vals[i] = __expf(vals[i] - mx);
        sum += vals[i];
    }
#pragma unroll
    for (int off = 8; off; off >>= 1) sum += __shfl_xor(sum, off, 16);
    float inv = __builtin_amdgcn_rcpf(sum);

    long rowbase = (long)(b * LQ + q0 + r) * LK;
    float*  arow = attn    + rowbase;
    __bf16* brow = attn_bf + rowbase;
#pragma unroll
    for (int i = 0; i < 16; ++i) {
        float w = vals[i] * inv;
        arow[j + 16 * i] = w;
        brow[j + 16 * i] = (__bf16)w;
    }
}

// ---------------------------------------------------------------------------
// Kernel 3: context = attn @ value  (bf16 WMMA; A=attn_bf, B=valueT)
//   One wave per 16x16 tile; K=256 fully unrolled -> 8 WMMAs/wave.
// ---------------------------------------------------------------------------
__global__ __launch_bounds__(256) void ctx_kernel(
    const __bf16* __restrict__ attn_bf, const __bf16* __restrict__ vT,
    float* __restrict__ ctx) {
    int wave = (blockIdx.x * 256 + threadIdx.x) >> 5;
    int lane = threadIdx.x & 31;
    int h = lane >> 4, ln = lane & 15;

    int b  = wave >> 9;       // 512 tiles per batch
    int t  = wave & 511;
    int tm = t >> 5;          // 16 M-tiles
    int tn = t & 31;          // 32 N-tiles

    const __bf16* arow = attn_bf + (long)b * LQ * LK + (long)(tm * 16 + ln) * LK;
    const __bf16* brow = vT      + (long)b * DV * LK + (long)(tn * 16 + ln) * LK;
    float*        C    = ctx     + (long)b * LQ * DV;

    v8f acc = {};
#pragma unroll
    for (int k0 = 0; k0 < LK; k0 += 32) {
        v16bf a = load_a_frag(arow, k0, h);
        v16bf bb = load_b_frag(brow, k0, h);
        acc = __builtin_amdgcn_wmma_f32_16x16x32_bf16(
            false, a, false, bb, (short)0, acc, false, false);
    }
#pragma unroll
    for (int r = 0; r < 8; ++r)
        C[(long)(tm * 16 + r + 8 * h) * DV + tn * 16 + ln] = acc[r];
}

// ---------------------------------------------------------------------------
extern "C" void kernel_launch(void* const* d_in, const int* in_sizes, int n_in,
                              void* d_out, int out_size, void* d_ws, size_t ws_size,
                              hipStream_t stream) {
    const float*         query = (const float*)d_in[0];
    const float*         key_t = (const float*)d_in[1];
    const float*         value = (const float*)d_in[2];
    const unsigned char* mask  = (const unsigned char*)d_in[3];
    const float*         Wq    = (const float*)d_in[4];
    const float*         Wk    = (const float*)d_in[5];
    const float*         Wv    = (const float*)d_in[6];

    float* out     = (float*)d_out;
    float* context = out;                          // B*LQ*DV floats
    float* attn    = out + (long)BB * LQ * DV;     // B*LQ*LK floats

    char* w = (char*)d_ws;
    float*  qp  = (float*)w;  w += (long)BB * LQ * UNI * 4;   // 2 MB
    float*  kp  = (float*)w;  w += (long)BB * LK * UNI * 4;   // 2 MB
    __bf16* qbf = (__bf16*)w; w += (long)BB * LQ * DQ * 2;    // 2 MB
    __bf16* kbf = (__bf16*)w; w += (long)BB * LK * DQ * 2;    // 2 MB
    __bf16* WqT = (__bf16*)w; w += (long)UNI * DQ * 2;        // 256 KB
    __bf16* WkT = (__bf16*)w; w += (long)UNI * DQ * 2;        // 256 KB
    __bf16* vT  = (__bf16*)w; w += (long)BB * DV * LK * 2;    // 2 MB
    __bf16* abf = (__bf16*)w; w += (long)BB * LQ * LK * 2;    // 1 MB

    int nQ = BB * LQ * DQ;   // 1,048,576
    cvt_bf16_kernel<<<nQ / 1024, 256, 0, stream>>>(query, qbf, nQ);
    cvt_bf16_kernel<<<nQ / 1024, 256, 0, stream>>>(key_t, kbf, nQ);
    // Wq/Wk: [512 x 256] -> [256 x 512] bf16
    transpose_cvt_kernel<<<dim3(UNI / 32, DQ / 32, 1), 256, 0, stream>>>(Wq, WqT, DQ, UNI);
    transpose_cvt_kernel<<<dim3(UNI / 32, DQ / 32, 1), 256, 0, stream>>>(Wk, WkT, DQ, UNI);
    // value: per batch [256 x 512] -> [512 x 256] bf16
    transpose_cvt_kernel<<<dim3(DV / 32, LK / 32, BB), 256, 0, stream>>>(value, vT, LK, DV);

    qk_proj_kernel<<<512, 256, 0, stream>>>(qbf, kbf, WqT, WkT, qp, kp);
    score_softmax_kernel<<<BB * (LQ / 16), 256, 0, stream>>>(qp, kp, Wv, mask, attn, abf);
    ctx_kernel<<<512, 256, 0, stream>>>(abf, vT, context);
}
